// MetaPathGNN_81252191306258
// MI455X (gfx1250) — matrix operand order repaired and checked by reading, not compiled, not run on hardware
//
#include <hip/hip_runtime.h>
#include <hip/hip_bf16.h>
#include <cstdint>
#include <cstddef>

// MetaPathGNN on MI455X (gfx1250), wave32, bf16 WMMA path.
//
// N=100000 nodes, C=256, OUT=128, E=400000 edges per edge-index.
//  - activations/weights in bf16 (51 MB activation matrix fits in 192 MB L2)
//  - aggregation: bf16 gathers + fp32 global atomics (bandwidth-bound part)
//  - 3 gate-folded GEMMs + bias + ReLU + LayerNorm fused in one WMMA kernel;
//    3 sources explicitly unrolled with distinct global pointers so A-loads
//    stay global_load_b128 (not flat), and per-tile B fragments live in
//    separate registers so load waits can stagger against WMMA issue.

#define C_DIM 256
#define OUT_DIM 128

typedef unsigned short u16;
typedef __attribute__((ext_vector_type(16))) __bf16 v16bf;
typedef __attribute__((ext_vector_type(8)))  __bf16 v8bf;
typedef __attribute__((ext_vector_type(8)))  float  v8f;

__device__ __forceinline__ u16 f2bf(float f) {
  unsigned u = __float_as_uint(f);
  u += 0x7FFFu + ((u >> 16) & 1u);   // round-to-nearest-even
  return (u16)(u >> 16);
}
__device__ __forceinline__ float bf2f(u16 h) {
  return __uint_as_float(((unsigned)h) << 16);
}

// ---------------- elementwise helpers ----------------

__global__ __launch_bounds__(256)
void k_cvt_bf16(const float* __restrict__ in, u16* __restrict__ out, int n) {
  int i = blockIdx.x * 256 + threadIdx.x;
  if (i < n) out[i] = f2bf(in[i]);
}

__global__ __launch_bounds__(256)
void k_deg(const int* __restrict__ dst, float* __restrict__ deg, int E) {
  int e = blockIdx.x * 256 + threadIdx.x;
  if (e < E) atomicAdd(&deg[dst[e]], 1.0f);
}

// one wave per edge; lane handles 8 channels (16B bf16 load, 8 fp32 atomics)
__global__ __launch_bounds__(256)
void k_scatter(const int* __restrict__ src, const int* __restrict__ dst,
               const u16* __restrict__ hb, float* __restrict__ agg, int E) {
  int wid  = (blockIdx.x * 256 + threadIdx.x) >> 5;
  int lane = threadIdx.x & 31;
  if (wid >= E) return;
  int s = src[wid], d = dst[wid];
  uint4 pk = *(const uint4*)(hb + (size_t)d * C_DIM + lane * 8);
  float* q = agg + (size_t)s * C_DIM + lane * 8;
  unsigned v[4] = {pk.x, pk.y, pk.z, pk.w};
#pragma unroll
  for (int j = 0; j < 4; ++j) {
    atomicAdd(&q[2 * j + 0], bf2f((u16)(v[j] & 0xffffu)));
    atomicAdd(&q[2 * j + 1], bf2f((u16)(v[j] >> 16)));
  }
}

__global__ __launch_bounds__(256)
void k_norm(const float* __restrict__ agg, const float* __restrict__ deg,
            u16* __restrict__ aggb, int total) {
  int i = blockIdx.x * 256 + threadIdx.x;
  if (i >= total) return;
  float d = deg[i >> 8];                 // C_DIM == 256
  d = d < 1.0f ? 1.0f : d;
  aggb[i] = f2bf(agg[i] / d);
}

// fold gate into weights: wbuf = [bf16(wl) ; bf16((1-g)w0) ; bf16(g*w1)],
// beff = bl + (1-g)b0 + g*b1,  g = sigmoid(gate)
__global__ __launch_bounds__(256)
void k_wprep(const float* __restrict__ wl, const float* __restrict__ w0,
             const float* __restrict__ w1, const float* __restrict__ bl,
             const float* __restrict__ b0, const float* __restrict__ b1,
             const float* __restrict__ gate, u16* __restrict__ wbuf,
             float* __restrict__ beff) {
  const float g = 1.0f / (1.0f + __expf(-gate[0]));
  const int CC = C_DIM * C_DIM;
  int i = blockIdx.x * 256 + threadIdx.x;
  if (i < CC) {
    wbuf[i]          = f2bf(wl[i]);
    wbuf[CC + i]     = f2bf((1.0f - g) * w0[i]);
    wbuf[2 * CC + i] = f2bf(g * w1[i]);
  }
  if (i < C_DIM) beff[i] = bl[i] + (1.0f - g) * b0[i] + g * b1[i];
}

// ---------------- WMMA K-loop over one (A, W) pair ----------------
// Accumulates NT 16x16 N-tiles. A fragment (16x32 bf16, ISA layout) reused
// across all NT tiles; the NT B fragments are loaded into separate register
// sets before the WMMAs so the compiler can stagger s_wait_loadcnt.
template <int NT>
__device__ __forceinline__ void wmma_k_accum(const u16* __restrict__ Arow,
                                             const u16* __restrict__ W,
                                             int ncol0, int lm, int hs,
                                             v8f* acc) {
#pragma unroll
  for (int k0 = 0; k0 < C_DIM; k0 += 32) {
    // A: lanes 0-15 -> K {k0..k0+7, k0+16..k0+23},
    //    lanes 16-31 -> K {k0+8..k0+15, k0+24..k0+31}
    v8bf alo = *(const v8bf*)(Arow + k0 + 8 * hs);
    v8bf ahi = *(const v8bf*)(Arow + k0 + 16 + 8 * hs);
    v16bf a = __builtin_shufflevector(alo, ahi, 0, 1, 2, 3, 4, 5, 6, 7, 8, 9,
                                      10, 11, 12, 13, 14, 15);
    v16bf b[NT];
#pragma unroll
    for (int nt = 0; nt < NT; ++nt) {
      const int n = ncol0 + nt * 16 + lm;   // lane holds row n of W
      v8bf blo = *(const v8bf*)(W + (size_t)n * C_DIM + k0 + 8 * hs);
      v8bf bhi = *(const v8bf*)(W + (size_t)n * C_DIM + k0 + 16 + 8 * hs);
      b[nt] = __builtin_shufflevector(blo, bhi, 0, 1, 2, 3, 4, 5, 6, 7, 8, 9,
                                      10, 11, 12, 13, 14, 15);
    }
#pragma unroll
    for (int nt = 0; nt < NT; ++nt) {
      acc[nt] = __builtin_amdgcn_wmma_f32_16x16x32_bf16(
          false, a, false, b[nt], (short)0, acc[nt], false, false);
    }
  }
}

// ---------------- fused 3xGEMM + bias + ReLU + LayerNorm ----------------
// Block = 128 threads (4 waves) owns rows [m0, m0+16), all 256 cols so the
// row-wise LayerNorm stays in-block. Wave wv covers cols [wv*64, wv*64+64).
__global__ __launch_bounds__(128)
void k_gemm_ln(const u16* __restrict__ aggb, const u16* __restrict__ hb,
               const u16* __restrict__ x0b, const u16* __restrict__ wbuf,
               const float* __restrict__ beff, const float* __restrict__ lns,
               const float* __restrict__ lnb, u16* __restrict__ outb) {
  const int m0   = blockIdx.x * 16;
  const int lane = threadIdx.x & 31;
  const int wv   = threadIdx.x >> 5;
  const int lm   = lane & 15;   // A row within tile / B,C column within tile
  const int hs   = lane >> 4;   // lane-group select (ISA fragment layout)
  const int ncol0 = wv * 64;

  v8f acc[4] = {};
  const size_t arow_off = (size_t)(m0 + lm) * C_DIM;
  const int CC = C_DIM * C_DIM;

  // three sources unrolled with distinct global pointers (keeps loads global)
  wmma_k_accum<4>(aggb + arow_off, wbuf,          ncol0, lm, hs, acc);
  wmma_k_accum<4>(hb   + arow_off, wbuf + CC,     ncol0, lm, hs, acc);
  wmma_k_accum<4>(x0b  + arow_off, wbuf + 2 * CC, ncol0, lm, hs, acc);

  // epilogue: bias + ReLU into LDS stripe, then row-wise LayerNorm
  __shared__ float tile[16][C_DIM];            // 16 KB of 320 KB
#pragma unroll
  for (int nt = 0; nt < 4; ++nt) {
    const int n = ncol0 + nt * 16 + lm;
    const float bia = beff[n];
#pragma unroll
    for (int r = 0; r < 8; ++r) {
      // C/D layout: element r of acc -> (M = r + 8*hs, N = lane%16)
      float v = acc[nt][r] + bia;
      tile[r + 8 * hs][n] = fmaxf(v, 0.0f);
    }
  }
  __syncthreads();

  __shared__ float psum[16][8], psq[16][8];
  {
    const int row = threadIdx.x & 15, seg = threadIdx.x >> 4;
    float s = 0.f, q = 0.f;
#pragma unroll
    for (int c = 0; c < 32; ++c) {
      float v = tile[row][seg * 32 + c];
      s += v;
      q += v * v;
    }
    psum[row][seg] = s;
    psq[row][seg]  = q;
  }
  __syncthreads();
  __shared__ float mu[16], rstd[16];
  if (threadIdx.x < 16) {
    float S = 0.f, Q = 0.f;
#pragma unroll
    for (int j = 0; j < 8; ++j) {
      S += psum[threadIdx.x][j];
      Q += psq[threadIdx.x][j];
    }
    float m = S * (1.0f / C_DIM);
    float v = Q * (1.0f / C_DIM) - m * m;
    mu[threadIdx.x]   = m;
    rstd[threadIdx.x] = rsqrtf(fmaxf(v, 0.0f) + 1e-5f);
  }
  __syncthreads();
  for (int i = threadIdx.x; i < 16 * C_DIM; i += 128) {
    int r = i >> 8, c = i & (C_DIM - 1);
    float v = (tile[r][c] - mu[r]) * rstd[r] * lns[c] + lnb[c];
    outb[(size_t)(m0 + r) * C_DIM + c] = f2bf(v);
  }
}

// ---------------- final projection: out = h2 @ Wout^T + bout ----------------
__global__ __launch_bounds__(128)
void k_gemm_out(const u16* __restrict__ hb, const u16* __restrict__ wb,
                const float* __restrict__ bout, float* __restrict__ out) {
  const int m0   = blockIdx.x * 16;
  const int lane = threadIdx.x & 31;
  const int wv   = threadIdx.x >> 5;
  const int lm   = lane & 15;
  const int hs   = lane >> 4;
  const int ncol0 = wv * 32;

  v8f acc[2] = {};
  wmma_k_accum<2>(hb + (size_t)(m0 + lm) * C_DIM, wb, ncol0, lm, hs, acc);

#pragma unroll
  for (int nt = 0; nt < 2; ++nt) {
    const int n  = ncol0 + nt * 16 + lm;
    const float bo = bout[n];
#pragma unroll
    for (int r = 0; r < 8; ++r) {
      out[(size_t)(m0 + r + 8 * hs) * OUT_DIM + n] = acc[nt][r] + bo;
    }
  }
}

// ---------------- launch ----------------

extern "C" void kernel_launch(void* const* d_in, const int* in_sizes, int n_in,
                              void* d_out, int out_size, void* d_ws,
                              size_t ws_size, hipStream_t stream) {
  (void)n_in; (void)out_size; (void)ws_size;

  const float* x     = (const float*)d_in[0];
  const int*   ei0   = (const int*)d_in[1];
  const int*   ei1   = (const int*)d_in[2];
  const float* wl0   = (const float*)d_in[3];
  const float* bl0   = (const float*)d_in[4];
  const float* w00   = (const float*)d_in[5];
  const float* b00   = (const float*)d_in[6];
  const float* w10   = (const float*)d_in[7];
  const float* b10   = (const float*)d_in[8];
  const float* gate0 = (const float*)d_in[9];
  const float* lns0  = (const float*)d_in[10];
  const float* lnb0  = (const float*)d_in[11];
  const float* wl1   = (const float*)d_in[12];
  const float* bl1   = (const float*)d_in[13];
  const float* w01   = (const float*)d_in[14];
  const float* b01   = (const float*)d_in[15];
  const float* w11   = (const float*)d_in[16];
  const float* b11   = (const float*)d_in[17];
  const float* gate1 = (const float*)d_in[18];
  const float* lns1  = (const float*)d_in[19];
  const float* lnb1  = (const float*)d_in[20];
  const float* Wout  = (const float*)d_in[21];
  const float* bout  = (const float*)d_in[22];

  const int N = in_sizes[0] / C_DIM;   // 100000
  const int E = in_sizes[1] / 2;       // 400000
  const size_t NC = (size_t)N * C_DIM;

  // workspace carve-out (~310 MB total)
  char* base = (char*)d_ws;
  size_t off = 0;
  auto take = [&](size_t bytes) -> void* {
    off = (off + 511) & ~(size_t)511;
    void* p = base + off;
    off += bytes;
    return p;
  };
  float* agg   = (float*)take(NC * 4);
  float* deg   = (float*)take((size_t)N * 4);
  u16*   xb    = (u16*)take(NC * 2);    // bf16(x) == bf16(x0): same tensor
  u16*   aggb  = (u16*)take(NC * 2);
  u16*   h1b   = (u16*)take(NC * 2);
  u16*   h2b   = (u16*)take(NC * 2);
  u16*   wbuf  = (u16*)take((size_t)3 * C_DIM * C_DIM * 2);
  float* beff  = (float*)take(C_DIM * 4);
  u16*   woutb = (u16*)take((size_t)OUT_DIM * C_DIM * 2);

  const int mtiles = N / 16;  // 6250, exact

  k_cvt_bf16<<<(int)((NC + 255) / 256), 256, 0, stream>>>(x, xb, (int)NC);
  k_cvt_bf16<<<(OUT_DIM * C_DIM + 255) / 256, 256, 0, stream>>>(
      Wout, woutb, OUT_DIM * C_DIM);

  auto run_layer = [&](const int* ei, const u16* hinb, const float* wl,
                       const float* bl, const float* w0, const float* b0,
                       const float* w1, const float* b1, const float* gate,
                       const float* lns, const float* lnb, u16* houtb) {
    hipMemsetAsync(agg, 0, NC * 4, stream);
    hipMemsetAsync(deg, 0, (size_t)N * 4, stream);
    k_deg<<<(E + 255) / 256, 256, 0, stream>>>(ei + E, deg, E);
    k_scatter<<<(E * 32 + 255) / 256, 256, 0, stream>>>(ei, ei + E, hinb, agg, E);
    k_norm<<<(int)((NC + 255) / 256), 256, 0, stream>>>(agg, deg, aggb, (int)NC);
    k_wprep<<<(C_DIM * C_DIM + 255) / 256, 256, 0, stream>>>(
        wl, w0, w1, bl, b0, b1, gate, wbuf, beff);
    k_gemm_ln<<<mtiles, 128, 0, stream>>>(aggb, hinb, xb, wbuf, beff, lns, lnb,
                                          houtb);
  };

  // metapath processed in reverse: conv index 1 first, then conv index 0
  run_layer(ei1, xb,  wl1, bl1, w01, b01, w11, b11, gate1, lns1, lnb1, h1b);
  run_layer(ei0, h1b, wl0, bl0, w00, b00, w10, b10, gate0, lns0, lnb0, h2b);

  k_gemm_out<<<mtiles, 128, 0, stream>>>(h2b, woutb, bout, (float*)d_out);
}
